// MixtureRouter_52707838657050
// MI455X (gfx1250) — matrix operand order repaired and checked by reading, not compiled, not run on hardware
//
#include <hip/hip_runtime.h>
#include <hip/hip_bf16.h>
#include <math.h>

typedef __attribute__((ext_vector_type(16))) __bf16 v16bf;
typedef __attribute__((ext_vector_type(8)))  __bf16 v8bf;
typedef __attribute__((ext_vector_type(8)))  float  v8f;

#define DDIM 2048
#define RDIM 512
#define BSZ  8
#define SDIM 2048
#define EXP  8
#define KCHUNK 256               // K staged per LDS chunk
#define NOUTER (DDIM / KCHUNK)   // 8 outer k-chunks
#define NKC (DDIM / 32)          // 64 total 32-wide k-steps
#define MTILE 64                 // rows per block (4 m-tiles of 16)

// ---------------------------------------------------------------------------
// Pack w1 (fp32, row-major [D,R]) into bf16 in this kernel's contiguous-k per-lane layout (same as A) of a
// V_WMMA B-matrix fragment (32x16, bf16): for fragment (nt, kc):
//   lane L (0..15):  column n = nt*16 + L,      holds K = kc*32 + 0..15
//   lane L (16..31): column n = nt*16 + (L-16), holds K = kc*32 + 16..31
// packed[((nt*64 + kc)*32 + lane)*16 + j]
// ---------------------------------------------------------------------------
__global__ __launch_bounds__(256) void pack_w1_bf16(const float* __restrict__ w1,
                                                    __bf16* __restrict__ w1p) {
  int o = blockIdx.x * 256 + threadIdx.x;          // 0 .. D*R-1 (exactly 1M)
  int j    =  o        & 15;
  int lane = (o >> 4)  & 31;
  int kc   = (o >> 9)  & 63;
  int nt   =  o >> 15;
  int k = kc * 32 + ((lane & 16) ? 16 : 0) + j;
  int n = nt * 16 + (lane & 15);
  w1p[o] = (__bf16)w1[k * RDIM + n];
}

// ---------------------------------------------------------------------------
// Fused: LayerNorm -> bf16 -> GEMM1 (WMMA bf16) -> +b1 -> GELU(exact) ->
// reduce over the 64 rows of this tile -> atomicAdd into hsum[B,R].
// Block = 256 threads (8 waves), handles 64 consecutive rows of [B*S, D].
// Wave w owns N-columns [w*64, w*64+64); all waves share all 4 m-tiles.
// ---------------------------------------------------------------------------
__global__ __launch_bounds__(256) void ln_gemm1_gelu_rowsum(
    const float* __restrict__ x, const float* __restrict__ gamma,
    const float* __restrict__ beta, const __bf16* __restrict__ w1p,
    const float* __restrict__ b1, float* __restrict__ hsum) {
  __shared__ __align__(16) __bf16 Atile[MTILE * KCHUNK];   // 32 KB normalized tile
  __shared__ float s_mean[MTILE], s_rstd[MTILE];

  const int tid  = threadIdx.x;
  const int wave = tid >> 5;
  const int lane = tid & 31;
  const long long rowBase = (long long)blockIdx.x * MTILE;
  const int b = blockIdx.x >> 5;                  // S/MTILE = 32 tiles per batch

  // ---- Pass 1: per-row mean / rstd (wave handles 8 rows, float4 loads) ----
  for (int rr = 0; rr < MTILE / 8; ++rr) {
    int m = wave * 8 + rr;
    const float4* rp = (const float4*)(x + (rowBase + m) * DDIM);
    float s = 0.f, s2 = 0.f;
    for (int i = lane; i < DDIM / 4; i += 32) {
      float4 v = rp[i];
      s  += v.x + v.y + v.z + v.w;
      s2 += v.x * v.x + v.y * v.y + v.z * v.z + v.w * v.w;
    }
    for (int off = 16; off > 0; off >>= 1) {
      s  += __shfl_xor(s, off);
      s2 += __shfl_xor(s2, off);
    }
    if (lane == 0) {
      float mu  = s / (float)DDIM;
      float var = s2 / (float)DDIM - mu * mu;
      s_mean[m] = mu;
      s_rstd[m] = rsqrtf(var + 1e-5f);
    }
  }
  __syncthreads();

  // accumulators: acc[i][j] = n-tile (wave*4+i) x m-tile j   (16x 16x16 tiles)
  v8f acc[4][4];
#pragma unroll
  for (int i = 0; i < 4; ++i)
#pragma unroll
    for (int j = 0; j < 4; ++j)
#pragma unroll
      for (int q = 0; q < 8; ++q) acc[i][j][q] = 0.f;

  // Per-thread B base pointer: fragment (i, kc) lives at
  //   bwave + ((i*NKC + kc) * 512) elements  -> compile-time offsets per (i,kk)
  const __bf16* bwave = w1p + ((size_t)(wave * 4) * NKC * 32 + (size_t)lane) * 16;
  // Per-thread A base offset inside LDS: frag (j, kk) at
  //   aoff + j*16*KCHUNK + kk*32 (+16 for second half) -> immediate ds offsets
  const int aoff = (lane & 15) * KCHUNK + ((lane & 16) ? 16 : 0);

  for (int c = 0; c < NOUTER; ++c) {
    // ---- stage normalized bf16 A chunk [MTILE][KCHUNK] into LDS ----
    {
      int m   = tid >> 2;                       // row 0..63 (4 threads per row)
      int seg = tid & 3;                        // 64-col segment
      int k0  = c * KCHUNK + seg * 64;
      const float4* rp = (const float4*)(x + (rowBase + m) * DDIM + k0);
      const float4* gp = (const float4*)(gamma + k0);
      const float4* bp = (const float4*)(beta + k0);
      float mu = s_mean[m], rs = s_rstd[m];
      __bf16* dst = &Atile[m * KCHUNK + seg * 64];
#pragma unroll
      for (int g = 0; g < 8; ++g) {             // 8 groups of 8 -> ds_store_b128
        float4 v0 = rp[g * 2],     v1 = rp[g * 2 + 1];
        float4 g0 = gp[g * 2],     g1 = gp[g * 2 + 1];
        float4 e0 = bp[g * 2],     e1 = bp[g * 2 + 1];
        v8bf t;
        t[0] = (__bf16)((v0.x - mu) * rs * g0.x + e0.x);
        t[1] = (__bf16)((v0.y - mu) * rs * g0.y + e0.y);
        t[2] = (__bf16)((v0.z - mu) * rs * g0.z + e0.z);
        t[3] = (__bf16)((v0.w - mu) * rs * g0.w + e0.w);
        t[4] = (__bf16)((v1.x - mu) * rs * g1.x + e1.x);
        t[5] = (__bf16)((v1.y - mu) * rs * g1.y + e1.y);
        t[6] = (__bf16)((v1.z - mu) * rs * g1.z + e1.z);
        t[7] = (__bf16)((v1.w - mu) * rs * g1.w + e1.w);
        *(v8bf*)&dst[g * 8] = t;
      }
    }
    // prefetch next chunk of this wave's packed weights
    if (c + 1 < NOUTER) {
      __builtin_prefetch(bwave + (size_t)(c + 1) * 8 * 512, 0, 1);  // global_prefetch_b8
    }
    __syncthreads();

    const __bf16* bc = bwave + (size_t)c * 8 * 512;    // kc = c*8 + kk
#pragma unroll
    for (int kk = 0; kk < 8; ++kk) {
      // A fragments for all 4 m-tiles (shared across waves, from LDS)
      v16bf afrag[4];
#pragma unroll
      for (int j = 0; j < 4; ++j) {
        v8bf a0 = *(const v8bf*)&Atile[aoff + j * 16 * KCHUNK + kk * 32];
        v8bf a1 = *(const v8bf*)&Atile[aoff + j * 16 * KCHUNK + kk * 32 + 8];
#pragma unroll
        for (int q = 0; q < 8; ++q) { afrag[j][q] = a0[q]; afrag[j][8 + q] = a1[q]; }
      }
#pragma unroll
      for (int i = 0; i < 4; ++i) {
        v16bf bfrag = *(const v16bf*)(bc + (size_t)(i * NKC + kk) * 512);
#pragma unroll
        for (int j = 0; j < 4; ++j) {
          acc[i][j] = __builtin_amdgcn_wmma_f32_16x16x32_bf16(
              false, afrag[j], false, bfrag, (short)0, acc[i][j], false, false);
        }
      }
    }
    __syncthreads();
  }

  // ---- epilogue: +b1, exact GELU, reduce all 64 rows, atomicAdd hsum[b][n] ----
#pragma unroll
  for (int i = 0; i < 4; ++i) {
    int n = (wave * 4 + i) * 16 + (lane & 15);
    float bias = b1[n];
    float s = 0.f;
#pragma unroll
    for (int j = 0; j < 4; ++j)
#pragma unroll
      for (int r = 0; r < 8; ++r) {
        float v = acc[i][j][r] + bias;                    // C: lane=N, vgpr=M
        s += 0.5f * v * (1.0f + erff(v * 0.70710678f));   // exact GELU
      }
    s += __shfl_xor(s, 16);                               // merge M halves
    if (lane < 16) atomicAdd(&hsum[b * RDIM + n], s);
  }
}

// ---------------------------------------------------------------------------
// Tail: batch_reduced = hsum @ w2 + S*b2 ; logits = batch_reduced @ wr + br ;
// BCE-with-logits loss vs one-hot(argmax) ; mode of top-1 indices.
// ---------------------------------------------------------------------------
__global__ __launch_bounds__(256) void finalize(
    const float* __restrict__ hsum, const float* __restrict__ w2,
    const float* __restrict__ b2, const float* __restrict__ wr,
    const float* __restrict__ br, float* __restrict__ out) {
  __shared__ float brd[BSZ * RDIM];     // 16 KB
  __shared__ float logits[BSZ * EXP];
  int tid = threadIdx.x;

  for (int idx = tid; idx < BSZ * RDIM; idx += 256) {
    int b = idx / RDIM, q = idx % RDIM;
    float acc = (float)SDIM * b2[q];
    for (int r = 0; r < RDIM; ++r) acc += hsum[b * RDIM + r] * w2[r * RDIM + q];
    brd[idx] = acc;
  }
  __syncthreads();

  if (tid < BSZ * EXP) {
    int b = tid / EXP, e = tid % EXP;
    float acc = br[e];
    for (int q = 0; q < RDIM; ++q) acc += brd[b * RDIM + q] * wr[q * EXP + e];
    logits[tid] = acc;
  }
  __syncthreads();

  if (tid == 0) {
    int counts[EXP];
    for (int e = 0; e < EXP; ++e) counts[e] = 0;
    float loss = 0.f;
    for (int b = 0; b < BSZ; ++b) {
      int am = 0;
      float mv = logits[b * EXP];
      for (int e = 1; e < EXP; ++e) {
        float v = logits[b * EXP + e];
        if (v > mv) { mv = v; am = e; }
      }
      counts[am]++;
      for (int e = 0; e < EXP; ++e) {
        float xv = logits[b * EXP + e];
        float sp = fmaxf(xv, 0.f) + log1pf(expf(-fabsf(xv)));  // logaddexp(0,x)
        loss += sp - (e == am ? xv : 0.f);
      }
    }
    loss /= (float)(BSZ * EXP);
    int ni = 0, mc = counts[0];
    for (int e = 1; e < EXP; ++e)
      if (counts[e] > mc) { mc = counts[e]; ni = e; }   // ties -> smallest idx
    out[0] = loss;
    out[1] = (float)ni;
  }
}

extern "C" void kernel_launch(void* const* d_in, const int* in_sizes, int n_in,
                              void* d_out, int out_size, void* d_ws, size_t ws_size,
                              hipStream_t stream) {
  const float* x     = (const float*)d_in[0];
  const float* gamma = (const float*)d_in[1];
  const float* beta  = (const float*)d_in[2];
  const float* w1    = (const float*)d_in[3];
  const float* b1    = (const float*)d_in[4];
  const float* w2    = (const float*)d_in[5];
  const float* b2    = (const float*)d_in[6];
  const float* wr    = (const float*)d_in[7];
  const float* br    = (const float*)d_in[8];
  float* out = (float*)d_out;

  __bf16* w1p  = (__bf16*)d_ws;                                    // 2 MB packed bf16
  float*  hsum = (float*)((char*)d_ws + (size_t)DDIM * RDIM * 2);  // 16 KB accumulator

  hipMemsetAsync(hsum, 0, (size_t)BSZ * RDIM * sizeof(float), stream);
  pack_w1_bf16<<<(DDIM * RDIM) / 256, 256, 0, stream>>>(w1, w1p);
  ln_gemm1_gelu_rowsum<<<(BSZ * SDIM) / MTILE, 256, 0, stream>>>(x, gamma, beta,
                                                                 w1p, b1, hsum);
  finalize<<<1, 256, 0, stream>>>(hsum, w2, b2, wr, br, out);
}
